// MaxMinSampler_11845519802582
// MI455X (gfx1250) — matrix operand order repaired and checked by reading, not compiled, not run on hardware
//
#include <hip/hip_runtime.h>
#include <hip/hip_fp16.h>

// Farthest Point Sampling for MI455X (gfx1250, wave32).
// Strategy: normalize once to f16 (32MB, L2-resident on a 192MB L2 part),
// then each of the 256 serial iterations is a [N,128]x[128] f16 WMMA mat-vec
// (query replicated across B's 16 columns) fused with s=max(s,dot) and a
// packed-u64 atomicMin argmin (value-encoded high, index low => exact
// first-index tie-break like jnp.argmax). A 1-block "pick" kernel between
// sweeps extracts the winner, poisons s[idx]=+inf, and stages the next query.
// The sweep loop is software-pipelined (double-buffered A fragments) so each
// tile's WMMAs overlap the next tile's L2 loads.

typedef __attribute__((ext_vector_type(16))) _Float16 v16h;
typedef __attribute__((ext_vector_type(8)))  _Float16 v8h;
typedef __attribute__((ext_vector_type(4)))  _Float16 v4h;
typedef __attribute__((ext_vector_type(8)))  float    v8f;

#define N_PTS   131072
#define DIM     128
#define K_SEL   256
#define N_TILES (N_PTS / 16)      // 8192 tiles of 16 points
#define UPD_BLOCKS 256            // 256 blocks x 8 waves = 2048 waves
#define TILES_PER_WAVE (N_TILES / (UPD_BLOCKS * 8))   // = 4

union H16 { v16h v; v8h h[2]; };

// Monotonic float->u32 map: f <= g  <=>  enc(f) <= enc(g) (unsigned).
static __device__ __forceinline__ unsigned enc_asc(float f) {
  unsigned b = __float_as_uint(f);
  return (b & 0x80000000u) ? ~b : (b | 0x80000000u);
}

// ---------------------------------------------------------------------------
// P0: L2-normalize rows (fp32 math, matching eps clamp) -> f16; init s=-inf;
// one warp (32 lanes) per row, 4 floats per lane.
__global__ __launch_bounds__(256) void fps_norm(
    const float* __restrict__ feat, _Float16* __restrict__ fnh,
    float* __restrict__ s, unsigned long long* __restrict__ slot)
{
  const int lane = threadIdx.x & 31;
  const int row  = blockIdx.x * (blockDim.x >> 5) + (threadIdx.x >> 5);
  const float4 f = ((const float4*)(feat + (size_t)row * DIM))[lane];
  float ss = f.x * f.x + f.y * f.y + f.z * f.z + f.w * f.w;
  #pragma unroll
  for (int o = 16; o > 0; o >>= 1) ss += __shfl_xor(ss, o, 32);
  const float inv = 1.0f / fmaxf(sqrtf(ss), 1e-12f);
  v4h h;
  h.x = (_Float16)(f.x * inv); h.y = (_Float16)(f.y * inv);
  h.z = (_Float16)(f.z * inv); h.w = (_Float16)(f.w * inv);
  *(v4h*)(fnh + (size_t)row * DIM + lane * 4) = h;
  if (lane == 0) s[row] = -__builtin_inff();
  if (row == 0 && lane == 0) *slot = ~0ULL;
}

// ---------------------------------------------------------------------------
// P1: first_idx = argmax(attention_scores), first index wins ties.
// Descending value encoding so atomicMin picks (max value, smallest index).
__global__ __launch_bounds__(256) void fps_first(
    const float* __restrict__ att, unsigned long long* __restrict__ slot)
{
  const int i = blockIdx.x * blockDim.x + threadIdx.x;
  const unsigned e = 0xFFFFFFFFu - enc_asc(att[i]);
  const unsigned long long cand =
      ((unsigned long long)e << 32) | (unsigned)i;
  __shared__ unsigned long long blk;
  if (threadIdx.x == 0) blk = ~0ULL;
  __syncthreads();
  atomicMin(&blk, cand);
  __syncthreads();
  if (threadIdx.x == 0) atomicMin(slot, blk);
}

// ---------------------------------------------------------------------------
// Issue the 8 b128 loads for one 16x128 A tile (per-lane fragment slices).
static __device__ __forceinline__ void load_tileA(
    const _Float16* __restrict__ arow, int khA,
    H16& a0, H16& a1, H16& a2, H16& a3)
{
  a0.h[0] = *(const v8h*)(arow +   0 + khA);
  a0.h[1] = *(const v8h*)(arow +  16 + khA);
  a1.h[0] = *(const v8h*)(arow +  32 + khA);
  a1.h[1] = *(const v8h*)(arow +  48 + khA);
  a2.h[0] = *(const v8h*)(arow +  64 + khA);
  a2.h[1] = *(const v8h*)(arow +  80 + khA);
  a3.h[0] = *(const v8h*)(arow +  96 + khA);
  a3.h[1] = *(const v8h*)(arow + 112 + khA);
}

// ---------------------------------------------------------------------------
// Update sweep: per wave, 4 tiles of 16 points. D = A(16x128 f16) x q via
// four chained v_wmma_f32_16x16x32_f16 (B = query replicated in all 16
// columns, preloaded once and pinned in VGPRs). Tile loop is double-buffered:
// next tile's loads issue before the pin-barrier on the current fragments, so
// the wait at the WMMAs is loadcnt<=8, not 0. s-updates deferred to epilogue
// so no scalar load forces loadcnt==0 inside the pipeline.
__global__ __launch_bounds__(256) void fps_update(
    const _Float16* __restrict__ fnh, const _Float16* __restrict__ q,
    float* __restrict__ s, unsigned long long* __restrict__ slot)
{
  const int lane = threadIdx.x & 31;
  const int wave = blockIdx.x * (blockDim.x >> 5) + (threadIdx.x >> 5);

  // B fragments: same K order as A — lanes 0-15 hold K {0..7,16..23},
  // lanes 16-31 hold K {8..15,24..31}. Columns identical => slices of q.
  const int khB = (lane >> 4) << 3;   // 0 or 8 (halves)
  H16 b0, b1, b2, b3;
  {
    const _Float16* qp = q + khB;
    b0.h[0] = *(const v8h*)(qp +  0); b0.h[1] = *(const v8h*)(qp +  16);
    b1.h[0] = *(const v8h*)(qp + 32); b1.h[1] = *(const v8h*)(qp +  48);
    b2.h[0] = *(const v8h*)(qp + 64); b2.h[1] = *(const v8h*)(qp +  80);
    b3.h[0] = *(const v8h*)(qp + 96); b3.h[1] = *(const v8h*)(qp + 112);
  }
  // Pin the query fragments in VGPRs (opaque to the optimizer: no re-loads).
  asm volatile("" : "+v"(b0.v), "+v"(b1.v), "+v"(b2.v), "+v"(b3.v));

  // A(16x32 f16) layout: lanes 0-15 = row m, K {0..7,16..23}; lanes 16-31
  // = row m, K {8..15,24..31}.
  const int khA  = (lane >> 4) << 3;  // 0 or 8 (halves)
  const int mrow = lane & 15;
  const int r    = lane & 7;
  const int tile0 = wave * TILES_PER_WAVE;

  // Prologue: load tile 0's fragments.
  H16 c0, c1, c2, c3;
  load_tileA(fnh + (size_t)(tile0 * 16 + mrow) * DIM, khA, c0, c1, c2, c3);

  float dv[TILES_PER_WAVE];
  #pragma unroll
  for (int t = 0; t < TILES_PER_WAVE; ++t) {
    // Issue next tile's loads BEFORE consuming the current fragments.
    H16 n0, n1, n2, n3;
    if (t + 1 < TILES_PER_WAVE) {
      load_tileA(fnh + (size_t)((tile0 + t + 1) * 16 + mrow) * DIM, khA,
                 n0, n1, n2, n3);
    }
    // Pin current fragments: wait only for their loads (next tile's 8 loads
    // remain outstanding -> s_wait_loadcnt 0x8, overlapping WMMA w/ L2).
    asm volatile("" : "+v"(c0.v), "+v"(c1.v), "+v"(c2.v), "+v"(c3.v));

    v8f acc = {};
    acc = __builtin_amdgcn_wmma_f32_16x16x32_f16(
        false, c0.v, false, b0.v, (short)0, acc, false, false);
    acc = __builtin_amdgcn_wmma_f32_16x16x32_f16(
        false, c1.v, false, b1.v, (short)0, acc, false, false);
    acc = __builtin_amdgcn_wmma_f32_16x16x32_f16(
        false, c2.v, false, b2.v, (short)0, acc, false, false);
    acc = __builtin_amdgcn_wmma_f32_16x16x32_f16(
        false, c3.v, false, b3.v, (short)0, acc, false, false);

    // All D columns equal: lanes 0-15 hold dots(rows 0-7) in acc[0..7],
    // lanes 16-31 hold dots(rows 8-15). Lanes {0-7,16-23} each own one row.
    float d = acc[0];
    d = (r == 1) ? acc[1] : d; d = (r == 2) ? acc[2] : d;
    d = (r == 3) ? acc[3] : d; d = (r == 4) ? acc[4] : d;
    d = (r == 5) ? acc[5] : d; d = (r == 6) ? acc[6] : d;
    d = (r == 7) ? acc[7] : d;
    dv[t] = d;                           // static index after full unroll

    if (t + 1 < TILES_PER_WAVE) {        // rotate double buffer
      c0 = n0; c1 = n1; c2 = n2; c3 = n3;
    }
  }

  // Epilogue: s = max(s, dot) + argmin candidates (outside the pipeline).
  unsigned long long best = ~0ULL;
  if ((lane & 8) == 0) {
    const int rowoff = r | ((lane >> 4) << 3);
    #pragma unroll
    for (int t = 0; t < TILES_PER_WAVE; ++t) {
      const int p = (tile0 + t) * 16 + rowoff;
      const float ns = fmaxf(s[p], dv[t]);   // selected points stay +inf
      s[p] = ns;
      const unsigned long long cand =
          ((unsigned long long)enc_asc(ns) << 32) | (unsigned)p;
      best = (cand < best) ? cand : best;
    }
  }

  __shared__ unsigned long long blk;
  if (threadIdx.x == 0) blk = ~0ULL;
  __syncthreads();
  atomicMin(&blk, best);
  __syncthreads();
  if (threadIdx.x == 0) atomicMin(slot, blk);
}

// ---------------------------------------------------------------------------
// Pick: read winner, emit index, poison s[idx], stage query, reset slot.
__global__ __launch_bounds__(128) void fps_pick(
    unsigned long long* __restrict__ slot, int* __restrict__ out, int iter,
    float* __restrict__ s, const _Float16* __restrict__ fnh,
    _Float16* __restrict__ q)
{
  const unsigned long long v = *slot;
  const int idx = (int)(unsigned)(v & 0xFFFFFFFFu);
  __syncthreads();                      // all reads of slot before reset
  if (threadIdx.x == 0) {
    out[iter] = idx;
    s[idx] = __builtin_inff();          // never re-selected by argmin
    *slot = ~0ULL;
  }
  if (threadIdx.x < 64) {               // 128 halves = 64 dwords
    ((unsigned*)q)[threadIdx.x] =
        ((const unsigned*)(fnh + (size_t)idx * DIM))[threadIdx.x];
  }
}

// ---------------------------------------------------------------------------
extern "C" void kernel_launch(void* const* d_in, const int* in_sizes, int n_in,
                              void* d_out, int out_size, void* d_ws, size_t ws_size,
                              hipStream_t stream) {
  const float* feat = (const float*)d_in[0];   // [131072,128] f32
  const float* att  = (const float*)d_in[1];   // [131072] f32
  (void)in_sizes; (void)n_in; (void)out_size;  // k=256 is compile-time

  int* out = (int*)d_out;                      // [256] int32 indices
  char* ws = (char*)d_ws; (void)ws_size;
  _Float16* fnh = (_Float16*)ws;                                   // 32 MB
  float*    s   = (float*)(ws + (size_t)N_PTS * DIM * 2);          // 512 KB
  _Float16* q   = (_Float16*)(ws + (size_t)N_PTS * DIM * 2
                                 + (size_t)N_PTS * 4);             // 256 B
  unsigned long long* slot =
      (unsigned long long*)(ws + (size_t)N_PTS * DIM * 2
                               + (size_t)N_PTS * 4 + 256);         // 8 B

  fps_norm <<<N_PTS / 8, 256, 0, stream>>>(feat, fnh, s, slot);
  fps_first<<<N_PTS / 256, 256, 0, stream>>>(att, slot);
  fps_pick <<<1, 128, 0, stream>>>(slot, out, 0, s, fnh, q);
  for (int it = 1; it < K_SEL; ++it) {
    fps_update<<<UPD_BLOCKS, 256, 0, stream>>>(fnh, q, s, slot);
    fps_pick  <<<1, 128, 0, stream>>>(slot, out, it, s, fnh, q);
  }
}
